// InterleavedHeadAttention_67920612819085
// MI455X (gfx1250) — matrix operand order repaired and checked by reading, not compiled, not run on hardware
//
#include <hip/hip_runtime.h>
#include <hip/hip_bf16.h>
#include <math.h>

// ---------------------------------------------------------------------------
// InterleavedHeadAttention for MI455X (gfx1250), bf16 WMMA path.
// B=2, S=1024, HID=1024, H=16, D=64, P=2, flat = S*P = 2048, BH = B*H = 32.
// Head-mix folded into projection weights:
//   W_eff[h,p,d,:] = sum_m alpha[m,h,p] * W[m*64+d,:]
// Workspace layout (total ~47 MiB):
//   [ 0MiB) Xbf      2048x1024 bf16    [ 4MiB) Wobf  1024x1024 bf16
//   [ 6MiB) Weff[3]  3x 2048x1024 bf16 [18MiB) beff[3] 3x 2048 f32
//   [19MiB) Q  (bh,flat,d) bf16        [27MiB) K (bh,flat,d) bf16
//   [35MiB) VT (bh,d,flat) bf16        [43MiB) AO 2048x1024 bf16
// ---------------------------------------------------------------------------

typedef __attribute__((ext_vector_type(16))) __bf16 v16bf;
typedef __attribute__((ext_vector_type(8)))  float  v8f;
typedef __attribute__((ext_vector_type(4)))  int    v4i;
typedef unsigned short BF16;

// b128 payload pointers in explicit address spaces (per builtin signature
// "vV4i*1V4i*3IiIi": global src, LDS dst, imm offset, imm cpol).
typedef __attribute__((address_space(1))) v4i AS1v4i;
typedef __attribute__((address_space(3))) v4i AS3v4i;

union FragBF { v16bf v; uint4 q[2]; };

__device__ __forceinline__ uint4 ld16(const BF16* p) {
  return *reinterpret_cast<const uint4*>(p);
}

__device__ __forceinline__ BF16 f2bf(float f) {
  unsigned int u = __float_as_uint(f);
  unsigned int r = u + 0x7FFFu + ((u >> 16) & 1u);  // round-to-nearest-even
  return (BF16)(r >> 16);
}

// ---- CDNA5 async global->LDS copy (ASYNCcnt path), with safe fallback -----
#if __has_builtin(__builtin_amdgcn_global_load_async_to_lds_b128)
#define HAVE_ASYNC_LDS 1
#endif

__device__ __forceinline__ void async_cp16(const BF16* g, BF16* l) {
#ifdef HAVE_ASYNC_LDS
  __builtin_amdgcn_global_load_async_to_lds_b128(
      (AS1v4i*)g, (AS3v4i*)l, 0, 0);
#else
  *reinterpret_cast<uint4*>(l) = *reinterpret_cast<const uint4*>(g);
#endif
}

__device__ __forceinline__ void wait_async0() {
#ifdef HAVE_ASYNC_LDS
#if __has_builtin(__builtin_amdgcn_s_wait_asynccnt)
  __builtin_amdgcn_s_wait_asynccnt(0);
#else
  asm volatile("s_wait_asynccnt 0" ::: "memory");
#endif
#endif
}

// wave-local LDS write->read ordering (no block barrier needed)
__device__ __forceinline__ void lds_fence() {
  asm volatile("s_wait_dscnt 0" ::: "memory");
}

// --------------------------- stage 0: casts --------------------------------
__global__ void __launch_bounds__(256)
cast_bf16_kernel(const float* __restrict__ src, BF16* __restrict__ dst, int n) {
  int i = blockIdx.x * 256 + threadIdx.x;
  if (i < n) dst[i] = f2bf(src[i]);
}

// --------------------- stage 1: effective weights --------------------------
__global__ void __launch_bounds__(256)
weff_kernel(const float* __restrict__ Wq, const float* __restrict__ bq,
            const float* __restrict__ Wk, const float* __restrict__ bk,
            const float* __restrict__ Wv, const float* __restrict__ bv,
            const float* __restrict__ aq, const float* __restrict__ ak,
            const float* __restrict__ av,
            BF16* __restrict__ weff, float* __restrict__ beff) {
  const int idx = blockIdx.x * 256 + threadIdx.x;     // 0 .. 2M-1
  const int j = idx >> 10;                            // output row (h,p,d)
  const int c = idx & 1023;                           // input channel
  const int h = j >> 7, p = (j >> 6) & 1, d = j & 63;

  const float *W, *bb, *al;
  if (blockIdx.y == 0)      { W = Wq; bb = bq; al = aq; }
  else if (blockIdx.y == 1) { W = Wk; bb = bk; al = ak; }
  else                      { W = Wv; bb = bv; al = av; }

  float s = 0.f;
  #pragma unroll
  for (int m = 0; m < 16; ++m)
    s += al[m * 32 + h * 2 + p] * W[(size_t)(m * 64 + d) * 1024 + c];
  weff[(size_t)blockIdx.y * (2048u * 1024u) + idx] = f2bf(s);

  if (c == 0) {
    float sb = 0.f;
    #pragma unroll
    for (int m = 0; m < 16; ++m)
      sb += al[m * 32 + h * 2 + p] * bb[m * 64 + d];
    beff[blockIdx.y * 2048 + j] = sb;
  }
}

// --------------------- stage 2/4: bf16 WMMA GEMM ---------------------------
// C[M,N] = A[M,K] * Bw[N,K]^T (+bias). One wave -> 32x64 tile, 8 accumulators,
// register double-buffered K-loop (loads for step k+1 overlap WMMA of step k).
// mode 0: bf16 out (bh,flat,d) (Q/K). mode 1: bf16 out (bh,d,flat) (V^T).
// mode 2: f32 out row-major + bias (final projection).

__device__ __forceinline__ void load_frags(const BF16* Arow0, const BF16* Arow1,
                                           const BF16* Brow0, int K, int ks,
                                           int ac0, int bc0,
                                           FragBF a[2], FragBF b[4]) {
  a[0].q[0] = ld16(Arow0 + ks + ac0);  a[0].q[1] = ld16(Arow0 + ks + ac0 + 16);
  a[1].q[0] = ld16(Arow1 + ks + ac0);  a[1].q[1] = ld16(Arow1 + ks + ac0 + 16);
  #pragma unroll
  for (int t = 0; t < 4; ++t) {
    const BF16* Br = Brow0 + (size_t)(t * 16) * K + ks + bc0;
    b[t].q[0] = ld16(Br);  b[t].q[1] = ld16(Br + 16);
  }
}

__device__ __forceinline__ void mma_step(const FragBF a[2], const FragBF b[4],
                                         v8f acc[2][4]) {
  #pragma unroll
  for (int mt = 0; mt < 2; ++mt)
    #pragma unroll
    for (int t = 0; t < 4; ++t)
      acc[mt][t] = __builtin_amdgcn_wmma_f32_16x16x32_bf16(
          false, a[mt].v, false, b[t].v, (short)0, acc[mt][t], false, false);
}

__global__ void __launch_bounds__(128)
gemm_bf16_kernel(const BF16* __restrict__ A, const BF16* __restrict__ Bw,
                 const float* __restrict__ bias,
                 BF16* __restrict__ out_bf, float* __restrict__ out_f,
                 int M, int N, int K, int mode) {
  const int lane  = threadIdx.x & 31;
  const int wave  = threadIdx.x >> 5;
  const int laneN = lane & 15;
  const int hi    = lane >> 4;
  const int Mbase = blockIdx.x * 32;
  const int Nbase = (blockIdx.y * 4 + wave) * 64;

  const BF16* Arow0 = A + (size_t)(Mbase + laneN) * K;
  const BF16* Arow1 = Arow0 + (size_t)16 * K;
  const BF16* Brow0 = Bw + (size_t)(Nbase + laneN) * K;
  const int ac0 = hi ? 8 : 0;          // A-frag K-chunk origin (ISA 16-bit A layout)
  const int bc0 = hi ? 8 : 0;          // B-frag K-chunk origin

  v8f acc[2][4] = {};
  FragBF a0[2], b0[4], a1[2], b1[4];

  load_frags(Arow0, Arow1, Brow0, K, 0, ac0, bc0, a0, b0);
  for (int ks = 0; ks < K; ks += 64) {
    __builtin_prefetch(Arow0 + ks + 128, 0, 0);   // global_prefetch_b8, 2 steps ahead
    __builtin_prefetch(Brow0 + ks + 128, 0, 0);
    if (ks + 32 < K) load_frags(Arow0, Arow1, Brow0, K, ks + 32, ac0, bc0, a1, b1);
    mma_step(a0, b0, acc);
    if (ks + 64 < K) load_frags(Arow0, Arow1, Brow0, K, ks + 64, ac0, bc0, a0, b0);
    if (ks + 32 < K) mma_step(a1, b1, acc);
  }

  // Epilogue. C layout: lane L, VGPR v -> row = base+v+8*(L>=16), col = Nbase+t*16+(L&15).
  #pragma unroll
  for (int mt = 0; mt < 2; ++mt) {
    #pragma unroll
    for (int t = 0; t < 4; ++t) {
      const int col = Nbase + t * 16 + laneN;
      #pragma unroll
      for (int v = 0; v < 8; ++v) {
        const int row = Mbase + mt * 16 + v + 8 * hi;
        float val = acc[mt][t][v] + bias[col];
        if (mode == 2) {
          out_f[(size_t)row * N + col] = val;
        } else {
          const int b = row >> 10, s = row & 1023;
          const int h = col >> 7, p = (col >> 6) & 1, d = col & 63;
          const int flat = s * 2 + p;
          const int bh = b * 16 + h;
          size_t off = (mode == 0)
                           ? ((size_t)bh * 2048 + flat) * 64 + d    // (bh, flat, d)
                           : ((size_t)bh * 64 + d) * 2048 + flat;   // (bh, d, flat)
          out_bf[off] = f2bf(val);
        }
      }
    }
  }
}

// --------------------- stage 3: flash attention ----------------------------
// Block = 4 waves = 64 q-rows of one (b,h). K/V tiles (32 keys) are staged into
// double-buffered LDS with async global->LDS copies and shared by all 4 waves.
// Token-causal: key tok <= query tok, tok = flat>>1. Causal skip is wave-uniform.
__global__ void __launch_bounds__(128)
attn_kernel(const BF16* __restrict__ Q, const BF16* __restrict__ Kt,
            const BF16* __restrict__ VT, const float* __restrict__ collapse,
            BF16* __restrict__ attnout) {
  __shared__ BF16 ldsK[2][32 * 64];   // [buf][key][d]   4 KB each
  __shared__ BF16 ldsV[2][64 * 32];   // [buf][d][key]   4 KB each
  __shared__ BF16 ldsP[4][16 * 32];   // per-wave P-tile staging (C->A relayout)

  const int tid   = threadIdx.x;
  const int wave  = tid >> 5;
  const int lane  = tid & 31;
  const int laneN = lane & 15;
  const int hi    = lane >> 4;
  const int qblk  = blockIdx.x * 64;
  const int qbase = qblk + wave * 16;
  const int bh    = blockIdx.y;
  const int b     = bh >> 4, h = bh & 15;

  const BF16* Qbh = Q  + (size_t)bh * 2048 * 64;
  const BF16* Kbh = Kt + (size_t)bh * 2048 * 64;
  const BF16* Vbh = VT + (size_t)bh * 64 * 2048;

  // Q A-fragments for d=[0,32) and [32,64)
  FragBF qf[2];
  {
    const BF16* qrow = Qbh + (size_t)(qbase + laneN) * 64;
    const int c0 = hi ? 8 : 0;
    qf[0].q[0] = ld16(qrow + c0);       qf[0].q[1] = ld16(qrow + c0 + 16);
    qf[1].q[0] = ld16(qrow + 32 + c0);  qf[1].q[1] = ld16(qrow + 32 + c0 + 16);
  }

  float mrow[8], lrow[8];
  v8f acc[4] = {};
  #pragma unroll
  for (int v = 0; v < 8; ++v) { mrow[v] = -__builtin_inff(); lrow[v] = 0.f; }

  const float sc = 0.125f * 1.44269504088896340736f;  // scale * log2(e)
  const int kend_blk = qblk + 64;    // max keys needed by any wave in block
  const int kend_wav = qbase + 16;   // this wave's causal bound

  for (int kb = 0; kb < kend_blk; kb += 32) {
    const int buf = (kb >> 5) & 1;

    // ---- stage K tile (waves 0-1) and V^T tile (waves 2-3) into LDS[buf]
    if (wave < 2) {
      // K tile is one contiguous 4 KB block: keys kb..kb+31, 64 d each.
      const BF16* src = Kbh + (size_t)kb * 64;
      #pragma unroll
      for (int c = 0; c < 4; ++c) {
        const int elem = ((wave * 4 + c) * 32 + lane) * 8;   // 8 bf16 = 16 B
        async_cp16(src + elem, &ldsK[buf][elem]);
      }
    } else {
      // V^T tile: 64 d-rows x 32 keys (64 B per row, global stride 4 KB).
      #pragma unroll
      for (int c = 0; c < 4; ++c) {
        const int idx = ((wave - 2) * 4 + c) * 32 + lane;    // 16 B chunk id
        const int d   = idx >> 2;
        const int ko  = (idx & 3) * 8;
        async_cp16(Vbh + (size_t)d * 2048 + kb + ko, &ldsV[buf][d * 32 + ko]);
      }
    }
    wait_async0();
    __syncthreads();   // uniform: every wave executes every iteration

    if (kb < kend_wav) {           // wave-uniform causal skip
      // ---- scores: two 16x16 tiles over the 32-key block (K from LDS)
      v8f sv[2];
      #pragma unroll
      for (int nt = 0; nt < 2; ++nt) {
        FragBF kf0, kf1;
        const BF16* krow = &ldsK[buf][(nt * 16 + laneN) * 64 + (hi ? 8 : 0)];
        kf0.q[0] = ld16(krow);       kf0.q[1] = ld16(krow + 16);
        kf1.q[0] = ld16(krow + 32);  kf1.q[1] = ld16(krow + 48);
        v8f s = {};
        s = __builtin_amdgcn_wmma_f32_16x16x32_bf16(false, qf[0].v, false, kf0.v,
                                                    (short)0, s, false, false);
        s = __builtin_amdgcn_wmma_f32_16x16x32_bf16(false, qf[1].v, false, kf1.v,
                                                    (short)0, s, false, false);
        sv[nt] = s;
      }

      // scale (log2 domain) + diagonal-block token-causal mask
      const bool needMask = (kb + 32 > qbase);
      #pragma unroll
      for (int nt = 0; nt < 2; ++nt) {
        const int ktok = (kb + nt * 16 + laneN) >> 1;
        #pragma unroll
        for (int v = 0; v < 8; ++v) {
          float x = sv[nt][v] * sc;
          if (needMask && ktok > ((qbase + v + 8 * hi) >> 1)) x = -__builtin_inff();
          sv[nt][v] = x;
        }
      }

      // ---- online softmax (rows live across 16-lane groups)
      #pragma unroll
      for (int v = 0; v < 8; ++v) {
        float rx = fmaxf(sv[0][v], sv[1][v]);
        #pragma unroll
        for (int m = 1; m < 16; m <<= 1) rx = fmaxf(rx, __shfl_xor(rx, m, 32));
        const float mnew = fmaxf(mrow[v], rx);
        const float al   = exp2f(mrow[v] - mnew);
        const float p0   = exp2f(sv[0][v] - mnew);
        const float p1   = exp2f(sv[1][v] - mnew);
        float rs = p0 + p1;
        #pragma unroll
        for (int m = 1; m < 16; m <<= 1) rs += __shfl_xor(rs, m, 32);
        lrow[v] = lrow[v] * al + rs;
        mrow[v] = mnew;
        #pragma unroll
        for (int t = 0; t < 4; ++t) acc[t][v] *= al;
        sv[0][v] = p0; sv[1][v] = p1;
      }

      // ---- P tile C->A relayout through this wave's private LDS slot
      #pragma unroll
      for (int v = 0; v < 8; ++v) {
        const int Mr = v + 8 * hi;
        ldsP[wave][Mr * 32 + laneN]      = f2bf(sv[0][v]);
        ldsP[wave][Mr * 32 + 16 + laneN] = f2bf(sv[1][v]);
      }
      lds_fence();   // same-wave LDS RAW; no block barrier (branch is non-uniform)
      FragBF pf;
      {
        const BF16* prow = &ldsP[wave][laneN * 32];
        const int c0 = hi ? 8 : 0;
        pf.q[0] = *reinterpret_cast<const uint4*>(prow + c0);
        pf.q[1] = *reinterpret_cast<const uint4*>(prow + c0 + 16);
      }

      // ---- P @ V from LDS (V^T gives contiguous keys per d-column)
      #pragma unroll
      for (int t = 0; t < 4; ++t) {
        FragBF vf;
        const BF16* vrow = &ldsV[buf][(t * 16 + laneN) * 32 + (hi ? 8 : 0)];
        vf.q[0] = ld16(vrow);
        vf.q[1] = ld16(vrow + 16);
        acc[t] = __builtin_amdgcn_wmma_f32_16x16x32_bf16(
            false, pf.v, false, vf.v, (short)0, acc[t], false, false);
      }
    }
  }

  // ---- normalize, collapse pseudo pairs (rows 2t, 2t+1), write bf16
  const float c0 = collapse[h * 2 + 0];
  const float c1 = collapse[h * 2 + 1];
  #pragma unroll
  for (int t = 0; t < 4; ++t) {
    const int col = h * 64 + t * 16 + laneN;
    #pragma unroll
    for (int pr = 0; pr < 4; ++pr) {
      const float o0 = acc[t][2 * pr]     / lrow[2 * pr];
      const float o1 = acc[t][2 * pr + 1] / lrow[2 * pr + 1];
      const int token = (qbase >> 1) + pr + 4 * hi;
      attnout[(size_t)(b * 1024 + token) * 1024 + col] = f2bf(c0 * o0 + c1 * o1);
    }
  }
}

// ---------------------------------------------------------------------------
extern "C" void kernel_launch(void* const* d_in, const int* in_sizes, int n_in,
                              void* d_out, int out_size, void* d_ws, size_t ws_size,
                              hipStream_t stream) {
  const float* hidden   = (const float*)d_in[0];
  const float* Wq       = (const float*)d_in[1];
  const float* bq       = (const float*)d_in[2];
  const float* Wk       = (const float*)d_in[3];
  const float* bk       = (const float*)d_in[4];
  const float* Wv       = (const float*)d_in[5];
  const float* bv       = (const float*)d_in[6];
  const float* Wo       = (const float*)d_in[7];
  const float* bo       = (const float*)d_in[8];
  const float* aq       = (const float*)d_in[9];
  const float* ak       = (const float*)d_in[10];
  const float* av       = (const float*)d_in[11];
  const float* collapse = (const float*)d_in[12];
  float* out            = (float*)d_out;

  char* ws = (char*)d_ws;
  const size_t MiB = 1u << 20;
  BF16*  Xbf  = (BF16*)(ws + 0 * MiB);
  BF16*  Wobf = (BF16*)(ws + 4 * MiB);
  BF16*  Weff = (BF16*)(ws + 6 * MiB);
  float* beff = (float*)(ws + 18 * MiB);
  BF16*  Qb   = (BF16*)(ws + 19 * MiB);
  BF16*  Kb   = (BF16*)(ws + 27 * MiB);
  BF16*  Vt   = (BF16*)(ws + 35 * MiB);
  BF16*  AO   = (BF16*)(ws + 43 * MiB);

  // stage 0: casts
  cast_bf16_kernel<<<(2048 * 1024) / 256, 256, 0, stream>>>(hidden, Xbf, 2048 * 1024);
  cast_bf16_kernel<<<(1024 * 1024) / 256, 256, 0, stream>>>(Wo, Wobf, 1024 * 1024);

  // stage 1: fold head-mix into projection weights
  weff_kernel<<<dim3((2048 * 1024) / 256, 3), 256, 0, stream>>>(
      Wq, bq, Wk, bk, Wv, bv, aq, ak, av, Weff, beff);

  // stage 2: Q, K (row layout) and V^T via WMMA GEMM (32x64 tile / wave)
  gemm_bf16_kernel<<<dim3(64, 8), 128, 0, stream>>>(
      Xbf, Weff + 0 * (size_t)2048 * 1024, beff + 0 * 2048, Qb, nullptr,
      2048, 2048, 1024, 0);
  gemm_bf16_kernel<<<dim3(64, 8), 128, 0, stream>>>(
      Xbf, Weff + 1 * (size_t)2048 * 1024, beff + 1 * 2048, Kb, nullptr,
      2048, 2048, 1024, 0);
  gemm_bf16_kernel<<<dim3(64, 8), 128, 0, stream>>>(
      Xbf, Weff + 2 * (size_t)2048 * 1024, beff + 2 * 2048, Vt, nullptr,
      2048, 2048, 1024, 1);

  // stage 3: flash attention + pseudo-head collapse (async LDS staging)
  attn_kernel<<<dim3(32, 32), 128, 0, stream>>>(Qb, Kb, Vt, collapse, AO);

  // stage 4: output projection (f32 + bias) into d_out
  gemm_bf16_kernel<<<dim3(64, 4), 128, 0, stream>>>(
      AO, Wobf, bo, nullptr, out, 2048, 1024, 1024, 2);
}
